// GraphConvolution_20117626815086
// MI455X (gfx1250) — matrix operand order repaired and checked by reading, not compiled, not run on hardware
//
#include <hip/hip_runtime.h>
#include <hip/hip_bf16.h>

typedef __attribute__((ext_vector_type(16))) _Float16 v16h;
typedef __attribute__((ext_vector_type(8)))  _Float16 v8h;
typedef __attribute__((ext_vector_type(4)))  _Float16 v4h;
typedef __attribute__((ext_vector_type(8)))  float    v8f;

#define LDA_PAD 40   // halves per LDS row (32-col tile, padded; 80B = 5x16B)

// ---------------------------------------------------------------------------
// Stage a 64x32 f32 tile (row stride src_stride) into LDS as f16, row-major
// with LDA_PAD stride. 256 threads: each converts 8 floats (two float4 loads,
// one b128 LDS store).
// ---------------------------------------------------------------------------
__device__ inline void stage_a64_f32_to_lds(const float* __restrict__ src,
                                            int src_stride, _Float16* dst,
                                            int tid) {
    int i   = tid >> 2;        // row 0..63
    int seg = (tid & 3) * 8;   // col 0,8,16,24
    const float4* p = (const float4*)(src + (size_t)i * src_stride + seg);
    float4 x = p[0];
    float4 y = p[1];
    v8h h;
    h[0] = (_Float16)x.x; h[1] = (_Float16)x.y;
    h[2] = (_Float16)x.z; h[3] = (_Float16)x.w;
    h[4] = (_Float16)y.x; h[5] = (_Float16)y.y;
    h[6] = (_Float16)y.z; h[7] = (_Float16)y.w;
    *(v8h*)(dst + i * LDA_PAD + seg) = h;
}

// ---------------------------------------------------------------------------
// Stage a 32x32 f32 tile into LDS as f16. 256 threads: each converts 4 floats
// (one float4 load, one b64 LDS store).
// ---------------------------------------------------------------------------
__device__ inline void stage_a32_f32_to_lds(const float* __restrict__ src,
                                            int src_stride, _Float16* dst,
                                            int tid) {
    int i   = tid >> 3;        // row 0..31
    int seg = (tid & 7) * 4;   // col 0,4,...,28
    float4 x = *(const float4*)(src + (size_t)i * src_stride + seg);
    v4h h;
    h[0] = (_Float16)x.x; h[1] = (_Float16)x.y;
    h[2] = (_Float16)x.z; h[3] = (_Float16)x.w;
    *(v4h*)(dst + i * LDA_PAD + seg) = h;
}

// ---------------------------------------------------------------------------
// A fragment (16x32 f16) from LDS, ISA 7.12.2 layout:
//   lane<16 : M=lane,    halves 0-7 = K0..7,  halves 8-15 = K16..23
//   lane>=16: M=lane-16, halves 0-7 = K8..15, halves 8-15 = K24..31
// ---------------------------------------------------------------------------
__device__ inline v16h load_frag_a(const _Float16* lds, int lane, int row_base) {
    int r  = row_base + (lane & 15);
    int kb = (lane >> 4) * 8;
    v8h lo = *(const v8h*)(lds + r * LDA_PAD + kb);
    v8h hi = *(const v8h*)(lds + r * LDA_PAD + kb + 16);
    return __builtin_shufflevector(lo, hi, 0,1,2,3,4,5,6,7,8,9,10,11,12,13,14,15);
}

// ---------------------------------------------------------------------------
// B fragment (32x16 f16) straight from a K-major (transposed) global matrix
// g[n][k] with row stride ld. ISA layout: lane holds column n0+lane%16,
// halves 0..7 = K = k0 + 8*(lane/16) + 0..7, halves 8..15 = K + 16 (two b128 loads).
// ---------------------------------------------------------------------------
__device__ inline v16h load_frag_b_global(const _Float16* __restrict__ g,
                                          int ld, int n0, int k0, int lane) {
    const _Float16* base = g + (size_t)(n0 + (lane & 15)) * ld
                             + k0 + ((lane >> 4) << 3);
    v8h lo = *(const v8h*)(base);
    v8h hi = *(const v8h*)(base + 16);
    return __builtin_shufflevector(lo, hi, 0,1,2,3,4,5,6,7,8,9,10,11,12,13,14,15);
}

// ---------------------------------------------------------------------------
// Kernel 1: degree -> s[i] = rsqrt(2 + sum_j adj[i][j])
// ---------------------------------------------------------------------------
__global__ __launch_bounds__(256) void gcn_degree_kernel(
        const float* __restrict__ adj, float* __restrict__ s) {
    __shared__ float red[256];
    int i = blockIdx.x, tid = threadIdx.x;
    const float4* row = (const float4*)(adj + (size_t)i * 8192);
    float sum = 0.f;
    #pragma unroll
    for (int c = 0; c < 8; ++c) {
        float4 v = row[tid + c * 256];
        sum += v.x + v.y + v.z + v.w;
    }
    red[tid] = sum;
    __syncthreads();
    for (int off = 128; off > 0; off >>= 1) {
        if (tid < off) red[tid] += red[tid + off];
        __syncthreads();
    }
    if (tid == 0) s[i] = rsqrtf(2.f + red[0]);
}

// ---------------------------------------------------------------------------
// Kernel 2: Wt[n][k] = f16(weight[k][n])   (512x256 -> 256x512)
// ---------------------------------------------------------------------------
__global__ __launch_bounds__(256) void gcn_wt_kernel(
        const float* __restrict__ w, _Float16* __restrict__ Wt) {
    int idx = blockIdx.x * 256 + threadIdx.x;   // 131072 total
    int k = idx >> 8;          // 0..511
    int n = idx & 255;         // 0..255  (coalesced read of w)
    Wt[(size_t)n * 512 + k] = (_Float16)w[idx];
}

// ---------------------------------------------------------------------------
// Kernel 3: T = s[j] * (input @ weight); stores T[j][n] and Tt[n][j] (f16).
// BM=64, BN=64, Kt=32. 8 waves as 2x4; wave tile 32x16.
// ---------------------------------------------------------------------------
__global__ __launch_bounds__(256) void gcn_support_kernel(
        const float* __restrict__ input, const _Float16* __restrict__ Wt,
        const float* __restrict__ s, _Float16* __restrict__ T,
        _Float16* __restrict__ Tt) {
    __shared__ __attribute__((aligned(16))) _Float16 Ah[2][64 * LDA_PAD];
    int tid  = threadIdx.x;
    int lane = tid & 31, wave = tid >> 5;
    int wrow = wave >> 2, wcol = wave & 3;
    int row0 = blockIdx.y * 64;
    int n0   = blockIdx.x * 64;

    v8f acc[2] = {};

    stage_a64_f32_to_lds(input + (size_t)row0 * 512, 512, Ah[0], tid);
    __syncthreads();

    const int NK = 512 / 32;
    for (int kt = 0; kt < NK; ++kt) {
        int cur = kt & 1;
        if (kt + 1 < NK)
            stage_a64_f32_to_lds(input + (size_t)row0 * 512 + (kt + 1) * 32,
                                 512, Ah[cur ^ 1], tid);
        int k0 = kt * 32;
        v16h b = load_frag_b_global(Wt, 512, n0 + wcol * 16, k0, lane);
        #pragma unroll
        for (int r = 0; r < 2; ++r) {
            v16h a = load_frag_a(Ah[cur], lane, wrow * 32 + r * 16);
            acc[r] = __builtin_amdgcn_wmma_f32_16x16x32_f16(
                false, a, false, b, (short)0, acc[r], false, false);
        }
        __syncthreads();
    }

    int mb = (lane >> 4) * 8;
    int nl = lane & 15;
    int n  = n0 + wcol * 16 + nl;
    #pragma unroll
    for (int r = 0; r < 2; ++r) {
        #pragma unroll
        for (int vr = 0; vr < 8; ++vr) {
            int i = row0 + wrow * 32 + r * 16 + mb + vr;
            float v = s[i] * acc[r][vr];
            _Float16 h = (_Float16)v;
            T [(size_t)i * 256  + n] = h;
            Tt[(size_t)n * 8192 + i] = h;
        }
    }
}

// ---------------------------------------------------------------------------
// Kernel 4: out = relu(2*s_i*T_i - s_i*(adj @ T)_i + b)
// BM=32 rows, BN=256 (full width -> adj streamed from HBM exactly once),
// Kt=32. 256 workgroups x 8 waves (2048 waves) for HBM latency hiding.
// Each wave owns a 32x32 output tile: 2x2 subtiles -> 4 wmma per K step.
// ---------------------------------------------------------------------------
__global__ __launch_bounds__(256) void gcn_aggregate_kernel(
        const float* __restrict__ adj, const _Float16* __restrict__ T,
        const _Float16* __restrict__ Tt, const float* __restrict__ s,
        const float* __restrict__ bias, float* __restrict__ out) {
    __shared__ __attribute__((aligned(16))) _Float16 Ah[2][32 * LDA_PAD];
    int tid  = threadIdx.x;
    int lane = tid & 31, wave = tid >> 5;   // wave 0..7 -> column slice
    int row0 = blockIdx.x * 32;

    v8f acc[2][2] = {};

    stage_a32_f32_to_lds(adj + (size_t)row0 * 8192, 8192, Ah[0], tid);
    __syncthreads();

    const int NK = 8192 / 32;
    for (int kt = 0; kt < NK; ++kt) {
        int cur = kt & 1;
        if (kt + 1 < NK)
            stage_a32_f32_to_lds(adj + (size_t)row0 * 8192 + (kt + 1) * 32,
                                 8192, Ah[cur ^ 1], tid);
        int k0 = kt * 32;
        v16h b[2];
        #pragma unroll
        for (int c = 0; c < 2; ++c)
            b[c] = load_frag_b_global(Tt, 8192, wave * 32 + c * 16, k0, lane);
        v16h a[2];
        #pragma unroll
        for (int r = 0; r < 2; ++r)
            a[r] = load_frag_a(Ah[cur], lane, r * 16);
        #pragma unroll
        for (int r = 0; r < 2; ++r)
            #pragma unroll
            for (int c = 0; c < 2; ++c)
                acc[r][c] = __builtin_amdgcn_wmma_f32_16x16x32_f16(
                    false, a[r], false, b[c], (short)0, acc[r][c], false, false);
        __syncthreads();
    }

    int mb = (lane >> 4) * 8;
    int nl = lane & 15;
    #pragma unroll
    for (int r = 0; r < 2; ++r) {
        #pragma unroll
        for (int c = 0; c < 2; ++c) {
            int n = wave * 32 + c * 16 + nl;
            float bn = bias[n];
            #pragma unroll
            for (int vr = 0; vr < 8; ++vr) {
                int i = row0 + r * 16 + mb + vr;
                float si = s[i];
                float tf = (float)T[(size_t)i * 256 + n];
                float v  = 2.f * si * tf - si * acc[r][c][vr] + bn;
                out[(size_t)i * 256 + n] = v > 0.f ? v : 0.f;
            }
        }
    }
}

// ---------------------------------------------------------------------------
extern "C" void kernel_launch(void* const* d_in, const int* in_sizes, int n_in,
                              void* d_out, int out_size, void* d_ws, size_t ws_size,
                              hipStream_t stream) {
    const float* input  = (const float*)d_in[0];   // 8192 x 512
    const float* adj    = (const float*)d_in[1];   // 8192 x 8192
    const float* weight = (const float*)d_in[2];   // 512 x 256
    const float* bias   = (const float*)d_in[3];   // 256
    float* out = (float*)d_out;                    // 8192 x 256

    char* ws = (char*)d_ws;
    float*    s  = (float*)ws;                               // 32 KB
    _Float16* T  = (_Float16*)(ws + 32 * 1024);              // 4 MB
    _Float16* Tt = T  + (size_t)8192 * 256;                  // 4 MB
    _Float16* Wt = Tt + (size_t)8192 * 256;                  // 256 KB

    gcn_degree_kernel<<<8192, 256, 0, stream>>>(adj, s);
    gcn_wt_kernel<<<512, 256, 0, stream>>>(weight, Wt);
    dim3 g2(4, 128);
    gcn_support_kernel<<<g2, 256, 0, stream>>>(input, Wt, s, T, Tt);
    gcn_aggregate_kernel<<<256, 256, 0, stream>>>(adj, T, Tt, s, bias, out);
}
